// CombinedAdaptiveMambaLayer_19834158973076
// MI455X (gfx1250) — matrix operand order repaired and checked by reading, not compiled, not run on hardware
//
#include <hip/hip_runtime.h>
#include <hip/hip_bf16.h>
#include <math.h>

// ---------------- constants ----------------
#define BB 4
#define CC 128
#define HH 64
#define WW 64
#define LL 4096            // H*W
#define HW 4096
#define D_INNER 256
#define D_STATE 16
#define DT_RANK 8
#define NXZ 512            // 2*D_INNER
#define NDBC_PAD 64        // 40 padded to 64 for GEMM N%32==0
#define NPATCH 256         // 16*16 patches of 4x4

typedef _Float16 v16h __attribute__((ext_vector_type(16)));
typedef float    v8f  __attribute__((ext_vector_type(8)));

__device__ __forceinline__ float sigmoidf_(float x) { return 1.f / (1.f + __expf(-x)); }
__device__ __forceinline__ float geluf_(float x)    { return 0.5f * x * (1.f + erff(x * 0.70710678118654752f)); }

// ---------------- K0: fp32 -> f16 weight prep (x_proj padded 40->64 rows) ---
__global__ __launch_bounds__(256) void k_prep_weights(
    const float* __restrict__ w_in, const float* __restrict__ w_xp,
    const float* __restrict__ w_out,
    _Float16* __restrict__ h_in, _Float16* __restrict__ h_xp,
    _Float16* __restrict__ h_out)
{
  int i = blockIdx.x * 256 + threadIdx.x;
  if (i < NXZ * CC)            h_in[i]  = (_Float16)w_in[i];
  if (i < NDBC_PAD * D_INNER)  h_xp[i]  = ((i >> 8) < 40) ? (_Float16)w_xp[i] : (_Float16)0.f;
  if (i < CC * D_INNER)        h_out[i] = (_Float16)w_out[i];
}

// ---------------- K1/K3: depthwise 3x3 SAME conv + GELU (+channel mean) -----
__global__ __launch_bounds__(256) void k_dwconv_gelu(
    const float* __restrict__ x, const float* __restrict__ w,
    const float* __restrict__ bias, const float* __restrict__ ca,
    float* __restrict__ y, float* __restrict__ ch_mean, int use_ca)
{
  int bc = blockIdx.x;                 // b*C + c
  int c  = bc & (CC - 1);
  const float* xin = x + (size_t)bc * HW;
  float scale = use_ca ? ca[bc] : 1.f;
  float w9[9];
#pragma unroll
  for (int i = 0; i < 9; ++i) w9[i] = w[c * 9 + i];
  float bv = bias[c];
  float lsum = 0.f;
  for (int p = threadIdx.x; p < HW; p += 256) {
    int hh = p >> 6, ww = p & 63;
    float acc = bv;
#pragma unroll
    for (int ky = 0; ky < 3; ++ky) {
      int yy = hh + ky - 1;
      if (yy < 0 || yy > 63) continue;
#pragma unroll
      for (int kx = 0; kx < 3; ++kx) {
        int xx = ww + kx - 1;
        if (xx < 0 || xx > 63) continue;
        acc += w9[ky * 3 + kx] * (scale * xin[yy * 64 + xx]);
      }
    }
    float g = geluf_(acc);
    y[(size_t)bc * HW + p] = g;
    lsum += g;
  }
  __shared__ float red[256];
  red[threadIdx.x] = lsum;
  __syncthreads();
  for (int o = 128; o > 0; o >>= 1) {
    if (threadIdx.x < o) red[threadIdx.x] += red[threadIdx.x + o];
    __syncthreads();
  }
  if (threadIdx.x == 0 && !use_ca) ch_mean[bc] = red[0] * (1.f / HW);
}

// ---------------- K2: channel-attention MLP -------------------------------
__global__ __launch_bounds__(128) void k_ca_mlp(
    const float* __restrict__ ca_mean, const float* __restrict__ w1,
    const float* __restrict__ w2, float* __restrict__ ca)
{
  int b = blockIdx.x, t = threadIdx.x;
  __shared__ float m[CC], hid[32];
  m[t] = ca_mean[b * CC + t];
  __syncthreads();
  if (t < 32) {
    float acc = 0.f;
    for (int cc2 = 0; cc2 < CC; ++cc2) acc += w1[t * CC + cc2] * m[cc2];
    hid[t] = fmaxf(acc, 0.f);
  }
  __syncthreads();
  float acc = 0.f;
#pragma unroll
  for (int j = 0; j < 32; ++j) acc += w2[t * 32 + j] * hid[j];
  ca[b * CC + t] = sigmoidf_(acc);
}

// ---------------- K4: per-pixel head score + channel variance --------------
__global__ __launch_bounds__(256) void k_scores(
    const float* __restrict__ x, const float* __restrict__ feat2,
    const float* __restrict__ head_w, const float* __restrict__ head_b,
    float* __restrict__ fs)
{
  int b = blockIdx.y;
  int p = blockIdx.x * 256 + threadIdx.x;     // 0..4095
  float dot = 0.f, sx = 0.f, sx2 = 0.f;
  for (int c = 0; c < CC; ++c) {
    size_t off = ((size_t)(b * CC + c)) * HW + p;
    dot += head_w[c] * feat2[off];
    float xv = x[off];
    sx += xv; sx2 += xv * xv;
  }
  float sc  = sigmoidf_(dot + head_b[0]);
  float mu  = sx * (1.f / CC);
  float pv  = (sx2 - (float)CC * mu * mu) * (1.f / (CC - 1));   // ddof=1
  fs[b * HW + p] = 0.7f * sc + 0.3f * sigmoidf_(pv * 10.f);
}

// ---------------- K5: patch means + bitonic argsort + order ----------------
__global__ __launch_bounds__(256) void k_sort_order(
    const float* __restrict__ fs, int* __restrict__ order)
{
  int b = blockIdx.x, t = threadIdx.x;
  __shared__ float val[NPATCH];
  __shared__ int   idx[NPATCH];
  int py = t >> 4, px = t & 15;
  float sum = 0.f;
#pragma unroll
  for (int oy = 0; oy < 4; ++oy)
#pragma unroll
    for (int ox = 0; ox < 4; ++ox)
      sum += fs[b * HW + (py * 4 + oy) * 64 + (px * 4 + ox)];
  val[t] = sum * (1.f / 16.f);
  idx[t] = t;
  __syncthreads();
  // descending by value, ties -> smaller patch index first (stable argsort(-sm))
  for (int k = 2; k <= NPATCH; k <<= 1) {
    for (int j = k >> 1; j > 0; j >>= 1) {
      int ixj = t ^ j;
      if (ixj > t) {
        bool dirAsc = ((t & k) == 0);
        float v0 = val[t], v1 = val[ixj];
        int   i0 = idx[t], i1 = idx[ixj];
        bool before01 = (v0 > v1) || (v0 == v1 && i0 < i1);
        bool sw = dirAsc ? !before01 : before01;
        if (sw) { val[t] = v1; val[ixj] = v0; idx[t] = i1; idx[ixj] = i0; }
      }
      __syncthreads();
    }
  }
  int p = idx[t];
  int ppy = p >> 4, ppx = p & 15;
#pragma unroll
  for (int jj = 0; jj < 16; ++jj) {
    int oy = jj >> 2, ox = jj & 3;
    order[b * LL + t * 16 + jj] = (ppy * 4 + oy) * 64 + (ppx * 4 + ox);
  }
}

// ---------------- K6: gather + positional embed + LayerNorm -> f16 ---------
__global__ __launch_bounds__(128) void k_gather_ln(
    const float* __restrict__ x, const float* __restrict__ pe,
    const float* __restrict__ g, const float* __restrict__ bta,
    const int* __restrict__ order, _Float16* __restrict__ x1h)
{
  int bl = blockIdx.x;                 // b*L + l
  int b  = bl >> 12;
  int c  = threadIdx.x;
  int pos = order[bl];
  float v = x[((size_t)(b * CC + c)) * HW + pos] + pe[(size_t)pos * CC + c];
  __shared__ float s1[CC], s2[CC];
  s1[c] = v; s2[c] = v * v;
  __syncthreads();
  for (int o = 64; o > 0; o >>= 1) {
    if (c < o) { s1[c] += s1[c + o]; s2[c] += s2[c + o]; }
    __syncthreads();
  }
  float mu  = s1[0] * (1.f / CC);
  float var = s2[0] * (1.f / CC) - mu * mu;
  float xn  = (v - mu) * rsqrtf(var + 1e-5f) * g[c] + bta[c];
  x1h[(size_t)bl * CC + c] = (_Float16)xn;
}

// ---------------- K7: shared f16 WMMA GEMM  Co[M,N] = A[M,K] * W[N,K]^T ----
// block: 256 thr = 8 waves; tile 128(M) x 32(N); each wave computes TWO 16x16
// M-tiles sharing one B fragment.  K stepped by 32 through LDS.
// Both A and B tiles are staged so every lane's fragment is 16 CONTIGUOUS
// halves -> lowers to ds_load_b128 pairs (no ds_load_u16 gather).
// Fragment k layouts used here (wave32, same order for A and B):
//   A 16x32 f16 : lanes 0-15 K 0..15,        lanes 16-31 K 16..31
//   B 32x16 f16 : lanes 0-15 K 0..15,        lanes 16-31 K 16..31
__global__ __launch_bounds__(256) void k_gemm_f16(
    const _Float16* __restrict__ A, const _Float16* __restrict__ Wn,
    float* __restrict__ Co, int M, int N, int K)
{
  __shared__ _Float16 As[128][40];   // 128x32 tile, rows padded to 40 halves
  __shared__ _Float16 Bl[32][40];    // row-major W tile: Bl[n][k]
  const int t    = threadIdx.x;
  const int wave = t >> 5, lane = t & 31;
  const int m0 = blockIdx.x * 128;
  const int n0 = blockIdx.y * 32;
  const int mt = (wave & 3) * 32;        // wave owns rows mt..mt+31 (2 tiles)
  const int nt = (wave >> 2) * 16;
  const int lhalf = lane & 15;
  const bool hi = lane >= 16;
  v8f acc0 = {0.f,0.f,0.f,0.f,0.f,0.f,0.f,0.f};
  v8f acc1 = {0.f,0.f,0.f,0.f,0.f,0.f,0.f,0.f};

  for (int k0 = 0; k0 < K; k0 += 32) {
    { // A tile: 128x32 halves = 4096; 16 halves/thread as two 16B loads
      int row = t >> 1;
      int cb  = (t & 1) * 16;
      const _Float16* src = A + (size_t)(m0 + row) * K + k0 + cb;
      *(float4*)(&As[row][cb])     = *(const float4*)(src);
      *(float4*)(&As[row][cb + 8]) = *(const float4*)(src + 8);
      if (k0 + 32 < K) __builtin_prefetch(src + 32, 0, 1);
    }
    if (t < 128) { // B tile: 32x32 halves = 1024; straight row copy
      int nn = t >> 2;
      int cb = (t & 3) * 8;
      *(float4*)(&Bl[nn][cb]) =
          *(const float4*)(Wn + (size_t)(n0 + nn) * K + k0 + cb);
    }
    __syncthreads();

    v16h a0, a1, b;
    const _Float16* ar0 = &As[mt + lhalf][0];
    const _Float16* ar1 = &As[mt + 16 + lhalf][0];
    const _Float16* br  = &Bl[nt + lhalf][0];
    const int ao = hi ? 16 : 0;
    const int bo = hi ? 16 : 0;
#pragma unroll
    for (int e = 0; e < 16; ++e) {
      int ka = ao + e;                        // contiguous 16 halves
      a0[e] = ar0[ka];
      a1[e] = ar1[ka];
      b[e]  = br[bo + e];                      // contiguous 16 halves
    }
    acc0 = __builtin_amdgcn_wmma_f32_16x16x32_f16(
        false, a0, false, b, (short)0, acc0, false, false);
    acc1 = __builtin_amdgcn_wmma_f32_16x16x32_f16(
        false, a1, false, b, (short)0, acc1, false, false);
    __syncthreads();
  }
#pragma unroll
  for (int r = 0; r < 8; ++r) {
    int rr  = hi ? r + 8 : r;
    int col = n0 + nt + lhalf;
    Co[(size_t)(m0 + mt + rr) * N + col]      = acc0[r];
    Co[(size_t)(m0 + mt + 16 + rr) * N + col] = acc1[r];
  }
}

// ---------------- K8: causal depthwise conv1d (k=4) + SiLU -----------------
__global__ __launch_bounds__(256) void k_conv1d(
    const float* __restrict__ xz, const float* __restrict__ w,
    const float* __restrict__ bias, float* __restrict__ xc,
    _Float16* __restrict__ xch)
{
  int bl = blockIdx.x;
  int l  = bl & (LL - 1);
  int d  = threadIdx.x;
  float acc = bias[d];
#pragma unroll
  for (int k = 0; k < 4; ++k) {
    int ll = l - 3 + k;
    if (ll >= 0) acc += w[d * 4 + k] * xz[(size_t)(bl - l + ll) * NXZ + d];
  }
  float s = acc * sigmoidf_(acc);          // silu
  xc [(size_t)bl * D_INNER + d] = s;
  xch[(size_t)bl * D_INNER + d] = (_Float16)s;
}

// ---------------- K9: dt = softplus(dt_low @ dt_proj^T + b) ----------------
__global__ __launch_bounds__(256) void k_dt(
    const float* __restrict__ dbc, const float* __restrict__ wdt,
    const float* __restrict__ bdt, float* __restrict__ dt)
{
  int bl = blockIdx.x;
  int d  = threadIdx.x;
  const float* row = dbc + (size_t)bl * NDBC_PAD;
  float acc = bdt[d];
#pragma unroll
  for (int r = 0; r < DT_RANK; ++r) acc += wdt[d * DT_RANK + r] * row[r];
  dt[(size_t)bl * D_INNER + d] = (acc > 20.f) ? acc : log1pf(__expf(acc));
}

// ---------------- K10: selective scan (sequential over L) ------------------
// wave32: lanes 0-15 = 16 states of channel d0, lanes 16-31 = channel d1.
__global__ __launch_bounds__(256) void k_scan(
    const float* __restrict__ dt, const float* __restrict__ xc,
    const float* __restrict__ dbc, const float* __restrict__ xz,
    const float* __restrict__ A_log, const float* __restrict__ Dp,
    _Float16* __restrict__ yh)
{
  int b    = blockIdx.x;
  int dblk = blockIdx.y;                       // 16 channels per block
  int wave = threadIdx.x >> 5, lane = threadIdx.x & 31;
  int d = dblk * 16 + wave * 2 + (lane >> 4);
  int s = lane & 15;
  float a  = -__expf(A_log[d * D_STATE + s]);
  float Dd = Dp[d];
  float h = 0.f;
  size_t base = (size_t)b * LL;
  for (int l = 0; l < LL; ++l) {
    size_t idx = base + l;
    float dtv = dt[idx * D_INNER + d];
    float xcv = xc[idx * D_INNER + d];
    float Bv  = dbc[idx * NDBC_PAD + DT_RANK + s];
    float Cv  = dbc[idx * NDBC_PAD + DT_RANK + D_STATE + s];
    h = __expf(dtv * a) * h + dtv * Bv * xcv;
    float p = h * Cv;
    p += __shfl_xor(p, 1, 32);
    p += __shfl_xor(p, 2, 32);
    p += __shfl_xor(p, 4, 32);
    p += __shfl_xor(p, 8, 32);               // sum within 16-lane half
    if (s == 0) {
      float z = xz[idx * NXZ + D_INNER + d];
      float y = (p + Dd * xcv) * (z * sigmoidf_(z));
      yh[idx * D_INNER + d] = (_Float16)y;
    }
  }
}

// ---------------- K11: scatter back via order ------------------------------
__global__ __launch_bounds__(256) void k_scatter(
    const float* __restrict__ outm, const int* __restrict__ order,
    float* __restrict__ out)
{
  int e = blockIdx.x * 256 + threadIdx.x;     // B*L*C
  if (e >= BB * LL * CC) return;
  int c  = e & (CC - 1);
  int bl = e >> 7;
  int b  = bl >> 12;
  int pos = order[bl];
  out[((size_t)(b * CC + c)) * HW + pos] = outm[(size_t)bl * CC + c];
}

// ======================= host side =========================================
static inline size_t align256(size_t x) { return (x + 255) & ~(size_t)255; }

extern "C" void kernel_launch(void* const* d_in, const int* in_sizes, int n_in,
                              void* d_out, int out_size, void* d_ws, size_t ws_size,
                              hipStream_t stream) {
  (void)in_sizes; (void)n_in; (void)out_size; (void)ws_size;
  const float* x        = (const float*)d_in[0];
  const float* pe       = (const float*)d_in[1];
  const float* norm_g   = (const float*)d_in[2];
  const float* norm_b   = (const float*)d_in[3];
  const float* conv1_w  = (const float*)d_in[4];
  const float* conv1_b  = (const float*)d_in[5];
  const float* ca1_w    = (const float*)d_in[6];
  const float* ca2_w    = (const float*)d_in[7];
  const float* conv2_w  = (const float*)d_in[8];
  const float* conv2_b  = (const float*)d_in[9];
  const float* head_w   = (const float*)d_in[10];
  const float* head_b   = (const float*)d_in[11];
  const float* in_proj  = (const float*)d_in[12];
  const float* conv1d_w = (const float*)d_in[13];
  const float* conv1d_b = (const float*)d_in[14];
  const float* x_proj_w = (const float*)d_in[15];
  const float* dt_pw    = (const float*)d_in[16];
  const float* dt_pb    = (const float*)d_in[17];
  const float* A_log    = (const float*)d_in[18];
  const float* Dp       = (const float*)d_in[19];
  const float* out_proj = (const float*)d_in[20];
  float* out = (float*)d_out;

  // workspace carve-up
  char* w = (char*)d_ws;
  size_t off = 0;
  auto take = [&](size_t bytes) { char* p = w + off; off += align256(bytes); return p; };
  float*     feat    = (float*)    take((size_t)BB * CC * HW * 4);
  float*     feat2   = (float*)    take((size_t)BB * CC * HW * 4);
  float*     ca_mean = (float*)    take((size_t)BB * CC * 4);
  float*     ca      = (float*)    take((size_t)BB * CC * 4);
  float*     fs      = (float*)    take((size_t)BB * HW * 4);
  int*       order   = (int*)      take((size_t)BB * LL * 4);
  _Float16*  x1h     = (_Float16*) take((size_t)BB * LL * CC * 2);
  _Float16*  h_in    = (_Float16*) take((size_t)NXZ * CC * 2);
  _Float16*  h_xp    = (_Float16*) take((size_t)NDBC_PAD * D_INNER * 2);
  _Float16*  h_out   = (_Float16*) take((size_t)CC * D_INNER * 2);
  float*     xz      = (float*)    take((size_t)BB * LL * NXZ * 4);
  float*     xc      = (float*)    take((size_t)BB * LL * D_INNER * 4);
  _Float16*  xch     = (_Float16*) take((size_t)BB * LL * D_INNER * 2);
  float*     dbc     = (float*)    take((size_t)BB * LL * NDBC_PAD * 4);
  float*     dt      = (float*)    take((size_t)BB * LL * D_INNER * 4);
  _Float16*  yh      = (_Float16*) take((size_t)BB * LL * D_INNER * 2);
  float*     outm    = (float*)    take((size_t)BB * LL * CC * 4);

  const int BL = BB * LL;   // 16384 GEMM rows

  // 0) weight conversion
  k_prep_weights<<<dim3(256), dim3(256), 0, stream>>>(in_proj, x_proj_w, out_proj,
                                                      h_in, h_xp, h_out);
  // 1) dwconv1 + gelu + channel means
  k_dwconv_gelu<<<dim3(BB * CC), dim3(256), 0, stream>>>(x, conv1_w, conv1_b, ca,
                                                         feat, ca_mean, 0);
  // 2) channel-attention MLP
  k_ca_mlp<<<dim3(BB), dim3(128), 0, stream>>>(ca_mean, ca1_w, ca2_w, ca);
  // 3) dwconv2 on (feat*ca) + gelu
  k_dwconv_gelu<<<dim3(BB * CC), dim3(256), 0, stream>>>(feat, conv2_w, conv2_b, ca,
                                                         feat2, ca_mean, 1);
  // 4) per-pixel scores + channel variance
  k_scores<<<dim3(HW / 256, BB), dim3(256), 0, stream>>>(x, feat2, head_w, head_b, fs);
  // 5) patch means + bitonic argsort -> pixel order
  k_sort_order<<<dim3(BB), dim3(256), 0, stream>>>(fs, order);
  // 6) gather + pe + LayerNorm -> f16 activations
  k_gather_ln<<<dim3(BL), dim3(128), 0, stream>>>(x, pe, norm_g, norm_b, order, x1h);
  // 7) in_proj GEMM: [16384,128] x [512,128]^T -> xz
  k_gemm_f16<<<dim3(BL / 128, NXZ / 32), dim3(256), 0, stream>>>(x1h, h_in, xz,
                                                                 BL, NXZ, CC);
  // 8) causal conv1d + silu
  k_conv1d<<<dim3(BL), dim3(D_INNER), 0, stream>>>(xz, conv1d_w, conv1d_b, xc, xch);
  // 9) x_proj GEMM: [16384,256] x [64(pad),256]^T -> dbc
  k_gemm_f16<<<dim3(BL / 128, NDBC_PAD / 32), dim3(256), 0, stream>>>(xch, h_xp, dbc,
                                                                      BL, NDBC_PAD, D_INNER);
  // 10) dt = softplus(...)
  k_dt<<<dim3(BL), dim3(D_INNER), 0, stream>>>(dbc, dt_pw, dt_pb, dt);
  // 11) selective scan (sequential over L, wave-parallel over states)
  k_scan<<<dim3(BB, D_INNER / 16), dim3(256), 0, stream>>>(dt, xc, dbc, xz,
                                                           A_log, Dp, yh);
  // 12) out_proj GEMM: [16384,256] x [128,256]^T -> outm
  k_gemm_f16<<<dim3(BL / 128, CC / 32), dim3(256), 0, stream>>>(yh, h_out, outm,
                                                                BL, CC, D_INNER);
  // 13) scatter back to (B,C,H,W)
  k_scatter<<<dim3((BB * LL * CC + 255) / 256), dim3(256), 0, stream>>>(outm, order, out);
}
